// BidirectionalAdditiveAttention_43997644981080
// MI455X (gfx1250) — matrix operand order repaired and checked by reading, not compiled, not run on hardware
//
#include <hip/hip_runtime.h>
#include <hip/hip_bf16.h>

// ---------------------------------------------------------------------------
// Problem constants (from reference): B=4, L1=L2=512, K1D=K2D=V1D=V2D=256, A=128
// d_out = o1[4,512,256] | o2[4,512,256] | w1[4,512,512] | w2[4,512,512] | score[4,512,512]
// ---------------------------------------------------------------------------

typedef __attribute__((ext_vector_type(16))) _Float16 v16h;
typedef __attribute__((ext_vector_type(8)))  float    v8f;

#define BB   4
#define LL   512
#define KD   256
#define AA   128
#define VD   256

// ------------------------------ math helpers -------------------------------

// Prefer the CDNA5 hardware tanh (v_tanh_f32, 1 TRANS op). Fallback:
// tanh(x) = 1 - 2/(exp(2x)+1) via v_exp_f32 + v_rcp_f32 (2 TRANS ops).
__device__ __forceinline__ float fast_tanh(float x) {
#if __has_builtin(__builtin_amdgcn_tanhf)
    return __builtin_amdgcn_tanhf(x);
#else
    float e = __builtin_amdgcn_exp2f(x * 2.8853900817779268f);  // exp(2x)
    return 1.0f - 2.0f * __builtin_amdgcn_rcpf(e + 1.0f);
#endif
}

__device__ __forceinline__ float fast_exp(float x) {
    return __builtin_amdgcn_exp2f(x * 1.4426950408889634f);
}

// --------------------------- WMMA tile loaders -----------------------------
// All leading dimensions are template parameters so every element address is
// base_ptr + compile-time immediate offset (folds into global_load IOFFSET).
//
// 16-bit A-matrix 16x32 k order used by these loaders (contiguous, same as B; not AMD's table order):
//   lane half h = lane>>4, m = lane&15
//   VGPR v (halves 2v,2v+1): k = 2v + e + h*16
template <int LDA>
__device__ __forceinline__ v16h load_A_f32(const float* __restrict__ src,
                                           int row0, int k0, int lane) {
    const int h = (lane >> 4) & 1;
    const int m = lane & 15;
    const float* base = src + (row0 + m) * LDA + k0 + h * 16;
    v16h a;
#pragma unroll
    for (int v = 0; v < 8; ++v) {
        const int kb = v * 2;   // compile-time
        a[2 * v + 0] = (_Float16)base[kb + 0];
        a[2 * v + 1] = (_Float16)base[kb + 1];
    }
    return a;
}

template <int LDA>
__device__ __forceinline__ v16h load_A_f16(const _Float16* __restrict__ src,
                                           int row0, int k0, int lane) {
    const int h = (lane >> 4) & 1;
    const int m = lane & 15;
    const _Float16* base = src + (row0 + m) * LDA + k0 + h * 16;
    v16h a;
#pragma unroll
    for (int v = 0; v < 8; ++v) {
        const int kb = v * 2;   // compile-time
        a[2 * v + 0] = base[kb + 0];
        a[2 * v + 1] = base[kb + 1];
    }
    return a;
}

// 16-bit B-matrix 32x16 layout: lanes 0-15 hold K=0..15 (N=lane),
// lanes 16-31 hold K=16..31 (N=lane-16); VGPR v holds K pair {2v,2v+1}.
template <int LDB>
__device__ __forceinline__ v16h load_B_f32(const float* __restrict__ src,
                                           int k0, int col0, int lane) {
    const int h = (lane >> 4) & 1;
    const int n = lane & 15;
    const float* base = src + (k0 + h * 16) * LDB + col0 + n;
    v16h b;
#pragma unroll
    for (int v = 0; v < 8; ++v) {
        b[2 * v + 0] = (_Float16)base[(v * 2 + 0) * LDB];  // imm offsets
        b[2 * v + 1] = (_Float16)base[(v * 2 + 1) * LDB];
    }
    return b;
}

template <int LDB>
__device__ __forceinline__ v16h load_B_f16(const _Float16* __restrict__ src,
                                           int k0, int col0, int lane) {
    const int h = (lane >> 4) & 1;
    const int n = lane & 15;
    const _Float16* base = src + (k0 + h * 16) * LDB + col0 + n;
    v16h b;
#pragma unroll
    for (int v = 0; v < 8; ++v) {
        b[2 * v + 0] = base[(v * 2 + 0) * LDB];            // imm offsets
        b[2 * v + 1] = base[(v * 2 + 1) * LDB];
    }
    return b;
}

// f32 C/D 16x16: VGPR r -> row r + 8*h, col = lane&15
template <int LDD>
__device__ __forceinline__ void store_D(float* __restrict__ dst,
                                        int row0, int col0, int lane, v8f d,
                                        float colBias) {
    const int h = (lane >> 4) & 1;
    const int n = lane & 15;
    float* base = dst + (row0 + 8 * h) * LDD + col0 + n;
#pragma unroll
    for (int r = 0; r < 8; ++r)
        base[r * LDD] = d[r] + colBias;                    // imm offsets
}

// ------------------------- 1) projection GEMM (WMMA) -----------------------
// P[M,N] = K[M,Kd] @ W[Kd,N] + bias[N]   (M=2048, Kd=256, N=128)
template <int Kd, int N>
__global__ void proj_wmma_kernel(const float* __restrict__ Kin,
                                 const float* __restrict__ W,
                                 const float* __restrict__ bias,
                                 float* __restrict__ P) {
    const int wave = (blockIdx.x * blockDim.x + threadIdx.x) >> 5;
    const int lane = threadIdx.x & 31;
    constexpr int ntn = N >> 4;
    const int tm = (wave / ntn) << 4;
    const int tn = (wave % ntn) << 4;

    v8f acc = {};
#pragma unroll 2
    for (int k0 = 0; k0 < Kd; k0 += 32) {
        v16h a = load_A_f32<Kd>(Kin, tm, k0, lane);
        v16h b = load_B_f32<N>(W, k0, tn, lane);
        acc = __builtin_amdgcn_wmma_f32_16x16x32_f16(
            false, a, false, b, (short)0, acc, false, false);
    }
    const float bv = bias[tn + (lane & 15)];
    store_D<N>(P, tm, tn, lane, acc, bv);
}

// ------------------------- 2) f32 -> f16 conversion ------------------------
__global__ void cvt_f16_kernel(const float* __restrict__ src,
                               _Float16* __restrict__ dst, int n4) {
    int i = blockIdx.x * blockDim.x + threadIdx.x;   // one float4 per thread
    if (i < n4) {
        const float4 v = ((const float4*)src)[i];
        v16h dummy;  (void)dummy;
        _Float16 o[4] = {(_Float16)v.x, (_Float16)v.y, (_Float16)v.z, (_Float16)v.w};
        *(uint2*)&dst[i * 4] = *(const uint2*)o;     // one b64 store
    }
}

// ------------------------- 3) additive score grid --------------------------
// score[b,i,j] = sum_a ws[a]*tanh(p1[b,i,a]+p2[b,j,a]) + bs, with pad mask.
// 64x64 (i,j) tile per 256-thread block, 4x4 register micro-tile per thread.
// p1/p2 tiles staged TRANSPOSED in LDS ([a][64], stride 68 for 16B alignment)
// so each inner iteration is 2x ds_load_b128 + 1 broadcast b32 for 16 tanh.
#define TS   64          // tile side
#define TP   68          // padded LDS row stride (floats), 16B-aligned
__global__ void score_kernel(const float* __restrict__ p1,
                             const float* __restrict__ p2,
                             const float* __restrict__ ws,
                             const float* __restrict__ bsp,
                             const int* __restrict__ len1,
                             const int* __restrict__ len2,
                             float* __restrict__ score) {
    const int b  = blockIdx.z;
    const int i0 = blockIdx.y * TS;
    const int j0 = blockIdx.x * TS;

    __shared__ __align__(16) float p1t[AA * TP];   // [a][i-local]
    __shared__ __align__(16) float p2t[AA * TP];   // [a][j-local]
    __shared__ float wss[AA];

    const int t = threadIdx.x;
    // stage transposed: thread x -> a = x&127 (coalesced global), r = x>>7
    for (int x = t; x < TS * AA; x += 256) {
        const int a = x & (AA - 1), r = x >> 7;
        p1t[a * TP + r] = p1[((b * LL) + i0 + r) * AA + a];
        p2t[a * TP + r] = p2[((b * LL) + j0 + r) * AA + a];
    }
    if (t < AA) wss[t] = ws[t];
    __syncthreads();

    const int ti = t >> 4, tj = t & 15;     // 16x16 thread grid of 4x4 micros
    const int ib = ti * 4, jb = tj * 4;

    float acc[4][4] = {};
#pragma unroll 2
    for (int a = 0; a < AA; ++a) {
        const float  w  = wss[a];
        const float4 xv = *(const float4*)&p1t[a * TP + ib];
        const float4 yv = *(const float4*)&p2t[a * TP + jb];
        const float xs[4] = {xv.x, xv.y, xv.z, xv.w};
        const float ys[4] = {yv.x, yv.y, yv.z, yv.w};
#pragma unroll
        for (int r = 0; r < 4; ++r)
#pragma unroll
            for (int c = 0; c < 4; ++c)
                acc[r][c] += w * fast_tanh(xs[r] + ys[c]);
    }

    const float bs  = bsp[0];
    const int   l1b = len1[b], l2b = len2[b];
    const int   jg  = j0 + jb;
    const int cp0 = (jg + 0 >= l2b), cp1 = (jg + 1 >= l2b);
    const int cp2 = (jg + 2 >= l2b), cp3 = (jg + 3 >= l2b);

#pragma unroll
    for (int r = 0; r < 4; ++r) {
        const int i  = i0 + ib + r;
        const int rp = (i >= l1b);
        const float ninf = -__builtin_inff();
        float4 o;
        o.x = ((rp + cp0) == 1) ? ninf : (acc[r][0] + bs);
        o.y = ((rp + cp1) == 1) ? ninf : (acc[r][1] + bs);
        o.z = ((rp + cp2) == 1) ? ninf : (acc[r][2] + bs);
        o.w = ((rp + cp3) == 1) ? ninf : (acc[r][3] + bs);
        *(float4*)&score[((b * LL) + i) * LL + jg] = o;
    }
}

// ------------------------- 4) softmax (rows & cols) ------------------------
// rows: w2[b,i,:] = softmax_j(score[b,i,:]);   one block per (b,i)
__global__ void softmax_rows_kernel(const float* __restrict__ score,
                                    float* __restrict__ wout,
                                    _Float16* __restrict__ wh) {
    const int b = blockIdx.y, i = blockIdx.x, t = threadIdx.x;
    const float* row = score + ((b * LL) + i) * LL;
    __shared__ float red[128];

    // 128 threads x one float4 each = 512 elements
    const float4 x = ((const float4*)row)[t];
    float m = fmaxf(fmaxf(x.x, x.y), fmaxf(x.z, x.w));
    red[t] = m;
    __syncthreads();
    for (int s = 64; s > 0; s >>= 1) {
        if (t < s) red[t] = fmaxf(red[t], red[t + s]);
        __syncthreads();
    }
    const float mx = red[0];
    __syncthreads();

    float4 e;
    e.x = fast_exp(x.x - mx); e.y = fast_exp(x.y - mx);
    e.z = fast_exp(x.z - mx); e.w = fast_exp(x.w - mx);
    red[t] = (e.x + e.y) + (e.z + e.w);
    __syncthreads();
    for (int s = 64; s > 0; s >>= 1) {
        if (t < s) red[t] = red[t] + red[t + s];
        __syncthreads();
    }
    const float inv = __builtin_amdgcn_rcpf(red[0]);

    float4 w4;
    w4.x = e.x * inv; w4.y = e.y * inv; w4.z = e.z * inv; w4.w = e.w * inv;
    ((float4*)(wout + ((b * LL) + i) * LL))[t] = w4;
    _Float16 h4[4] = {(_Float16)w4.x, (_Float16)w4.y, (_Float16)w4.z, (_Float16)w4.w};
    *(uint2*)&wh[((b * LL) + i) * LL + t * 4] = *(const uint2*)h4;
}

// cols: w1[b,j,:] = softmax_i(score[b,:,j]);   one block per (b,j)
__global__ void softmax_cols_kernel(const float* __restrict__ score,
                                    float* __restrict__ wout,
                                    _Float16* __restrict__ wh) {
    const int b = blockIdx.y, j = blockIdx.x, t = threadIdx.x;
    const float* base = score + (b * LL) * LL + j;
    __shared__ float red[256];

    float x0 = base[t * LL], x1 = base[(t + 256) * LL];
    red[t] = fmaxf(x0, x1);
    __syncthreads();
    for (int s = 128; s > 0; s >>= 1) {
        if (t < s) red[t] = fmaxf(red[t], red[t + s]);
        __syncthreads();
    }
    const float mx = red[0];
    __syncthreads();

    const float e0 = fast_exp(x0 - mx), e1 = fast_exp(x1 - mx);
    red[t] = e0 + e1;
    __syncthreads();
    for (int s = 128; s > 0; s >>= 1) {
        if (t < s) red[t] = red[t] + red[t + s];
        __syncthreads();
    }
    const float inv = __builtin_amdgcn_rcpf(red[0]);

    const float w0 = e0 * inv, w1 = e1 * inv;
    float* o = wout + ((b * LL) + j) * LL;
    _Float16* oh = wh + ((b * LL) + j) * LL;
    o[t] = w0;            o[t + 256] = w1;
    oh[t] = (_Float16)w0; oh[t + 256] = (_Float16)w1;
}

// ------------------------- 5) output GEMMs (WMMA) --------------------------
// O[b] = Wm[b] (f16, [M,K]) @ Vm[b] (f16, [K,N]) ; M=512,K=512,N=256
template <int Kd, int N>
__global__ void out_gemm_kernel(const _Float16* __restrict__ Wm,
                                const _Float16* __restrict__ Vm,
                                float* __restrict__ O, int M) {
    const int b = blockIdx.y;
    const _Float16* Ab = Wm + (size_t)b * M * Kd;
    const _Float16* Bb = Vm + (size_t)b * Kd * N;
    float* Ob = O + (size_t)b * M * N;

    const int wave = (blockIdx.x * blockDim.x + threadIdx.x) >> 5;
    const int lane = threadIdx.x & 31;
    constexpr int ntn = N >> 4;
    const int tm = (wave / ntn) << 4;
    const int tn = (wave % ntn) << 4;

    v8f acc = {};
#pragma unroll 2
    for (int k0 = 0; k0 < Kd; k0 += 32) {
        v16h a  = load_A_f16<Kd>(Ab, tm, k0, lane);
        v16h bt = load_B_f16<N>(Bb, k0, tn, lane);
        acc = __builtin_amdgcn_wmma_f32_16x16x32_f16(
            false, a, false, bt, (short)0, acc, false, false);
    }
    store_D<N>(Ob, tm, tn, lane, acc, 0.0f);
}

// ------------------------------ launcher -----------------------------------
extern "C" void kernel_launch(void* const* d_in, const int* in_sizes, int n_in,
                              void* d_out, int out_size, void* d_ws, size_t ws_size,
                              hipStream_t stream) {
    const float* k1   = (const float*)d_in[0];
    const float* k2   = (const float*)d_in[1];
    const float* v1   = (const float*)d_in[2];
    const float* v2   = (const float*)d_in[3];
    const float* W1   = (const float*)d_in[4];
    const float* b1   = (const float*)d_in[5];
    const float* W2   = (const float*)d_in[6];
    const float* b2   = (const float*)d_in[7];
    const float* ws   = (const float*)d_in[8];
    const float* bs   = (const float*)d_in[9];
    const int*   len1 = (const int*)d_in[10];
    const int*   len2 = (const int*)d_in[11];

    const int NP  = BB * LL * AA;      // 262144  p1/p2 elements
    const int NV  = BB * LL * VD;      // 524288  v elements
    const int NW  = BB * LL * LL;      // 1048576 w/score elements

    // workspace: p1,p2 (f32) | v1h,v2h (f16) | w1h,w2h (f16)  => 8 MB total
    float*     p1  = (float*)d_ws;
    float*     p2  = p1 + NP;
    _Float16*  v1h = (_Float16*)(p2 + NP);
    _Float16*  v2h = v1h + NV;
    _Float16*  w1h = v2h + NV;
    _Float16*  w2h = w1h + NW;

    // outputs: o1 | o2 | w1 | w2 | score
    float* o1  = (float*)d_out;
    float* o2  = o1 + NV;
    float* w1o = o2 + NV;
    float* w2o = w1o + NW;
    float* sc  = w2o + NW;

    // 1) projections: [2048,256]@[256,128]+bias -> p1/p2 (1024 wave-tiles each)
    proj_wmma_kernel<KD, AA><<<128, 256, 0, stream>>>(k1, W1, b1, p1);
    proj_wmma_kernel<KD, AA><<<128, 256, 0, stream>>>(k2, W2, b2, p2);

    // 2) v1/v2 -> f16 (float4 per thread)
    cvt_f16_kernel<<<(NV / 4 + 255) / 256, 256, 0, stream>>>(v1, v1h, NV / 4);
    cvt_f16_kernel<<<(NV / 4 + 255) / 256, 256, 0, stream>>>(v2, v2h, NV / 4);

    // 3) score grid with mask (+bs): 64x64 tiles
    {
        dim3 g(LL / TS, LL / TS, BB);   // 8 x 8 x 4
        score_kernel<<<g, 256, 0, stream>>>(p1, p2, ws, bs, len1, len2, sc);
    }

    // 4) softmaxes: w2 = rows, w1 = cols (f32 to d_out, f16 to ws)
    {
        dim3 g(LL, BB);
        softmax_rows_kernel<<<g, 128, 0, stream>>>(sc, w2o, w2h);
        softmax_cols_kernel<<<g, 256, 0, stream>>>(sc, w1o, w1h);
    }

    // 5) output GEMMs: o1 = w1 @ v1, o2 = w2 @ v2  (512 wave-tiles per batch)
    {
        dim3 g(64, BB);
        out_gemm_kernel<LL, VD><<<g, 256, 0, stream>>>(w1h, v1h, o1, LL);
        out_gemm_kernel<LL, VD><<<g, 256, 0, stream>>>(w2h, v2h, o2, LL);
    }
}
